// SelfAttention_5128190952087
// MI455X (gfx1250) — matrix operand order repaired and checked by reading, not compiled, not run on hardware
//
#include <hip/hip_runtime.h>
#include <hip/hip_bf16.h>
#include <math.h>

// Self-attention (SAGAN) for B=4, C=576, N=4096, Cqk=64 on gfx1250.
// Pipeline: 3x projection GEMM (bf16 WMMA) -> softmax stats -> fused PV + residual.
// attn_out stages the value tile via the Tensor Data Mover (tensor_load_to_lds),
// with h kept in natural [C][N] layout so PV B-fragments are contiguous LDS reads.
// Workspace layout (bytes):
//   fT (keys^T)   [4][4096][64]  bf16 @ 0          (2 MiB)
//   gT (queries^T)[4][4096][64]  bf16 @ 2 MiB      (2 MiB)
//   hN (values)   [4][576][4096] bf16 @ 4 MiB      (18 MiB)
//   Mst           [4][4096]      f32  @ 4MiB+18874368
//   Lst           [4][4096]      f32  @ +65536     (total ~22.2 MiB)

typedef __bf16 bf16_t;
typedef __attribute__((ext_vector_type(16))) __bf16 v16bf;
typedef __attribute__((ext_vector_type(8)))  __bf16 v8bf;
typedef __attribute__((ext_vector_type(8)))  float  v8f;
typedef __attribute__((ext_vector_type(4)))  unsigned int v4u;
typedef __attribute__((ext_vector_type(8)))  int v8i;
typedef __attribute__((ext_vector_type(4)))  int v4i;

#define WMMA_BF16(a, b, c) \
  __builtin_amdgcn_wmma_f32_16x16x32_bf16(false, (a), false, (b), (short)0, (c), false, false)

// A-operand fragment (16x32 bf16, row-major source row):
// lane<16 holds K {0..15}; lane>=16 holds K {16..31}  (kb = (lane>>4)*8)
static __device__ inline v16bf load_afrag(const bf16_t* __restrict__ row, int kb) {
  v8bf lo = *(const v8bf*)(row + 2 * kb);
  v8bf hi = *(const v8bf*)(row + 2 * kb + 8);
  v16bf r;
#pragma unroll
  for (int i = 0; i < 8; ++i) { r[i] = lo[i]; r[i + 8] = hi[i]; }
  return r;
}

// Tensor Data Mover: DMA a 2D bf16 tile [tile_d1 rows x tile_d0 cols] out of a
// [tensor_d1 x tensor_d0] row-major tensor (row stride = stride0 elements) into
// LDS at lds_off (tile packed row-major [tile_d1][tile_d0]).
// D# built per CDNA5 ISA ch.8: group0 = {count/type/addr}, group1 = dims/strides.
static __device__ inline void tdm_load_2d_bf16(unsigned lds_off, const void* gaddr,
                                               unsigned tile_d0, unsigned tile_d1,
                                               unsigned tensor_d0, unsigned tensor_d1,
                                               unsigned long long stride0) {
  unsigned long long ga = (unsigned long long)(uintptr_t)gaddr;
  v4u g0;
  g0[0] = 1u;                                                // count=1, user mode
  g0[1] = lds_off;                                           // lds_addr (bytes)
  g0[2] = (unsigned)(ga & 0xFFFFFFFFu);                      // global_addr[31:0]
  g0[3] = (unsigned)((ga >> 32) & 0x01FFFFFFu) | (2u << 30); // addr[56:32], type=2
  v8i g1;
  g1[0] = (int)(1u << 16);                                   // data_size=1 (2B), mask=0
  g1[1] = (int)((tensor_d0 & 0xFFFFu) << 16);                // tensor_dim0[15:0]
  g1[2] = (int)((tensor_d0 >> 16) | ((tensor_d1 & 0xFFFFu) << 16));
  g1[3] = (int)((tensor_d1 >> 16) | ((tile_d0 & 0xFFFFu) << 16)); // dim1 hi | tile_dim0
  g1[4] = (int)(tile_d1 & 0xFFFFu);                          // tile_dim1; tile_dim2=0
  g1[5] = (int)(stride0 & 0xFFFFFFFFu);                      // tensor_dim0_stride lo
  g1[6] = (int)((stride0 >> 32) & 0xFFFFu);                  // stride0 hi; stride1=0
  g1[7] = 0;
  v4i g2 = {0, 0, 0, 0};
  v4i g3 = {0, 0, 0, 0};
#if __clang_major__ >= 23
  v8i g4 = {0, 0, 0, 0, 0, 0, 0, 0};
  __builtin_amdgcn_tensor_load_to_lds(g0, g1, g2, g3, g4, 0);
#else
  __builtin_amdgcn_tensor_load_to_lds(g0, g1, g2, g3, 0);
#endif
}

// ---------------------------------------------------------------------------
// Kernel 1: P[b] = W[Mout,576] * X[b][576,4096] in bf16 WMMA.
// transposed!=0 -> store out^T[b][n][Mout] (A-fragment friendly, 16B stores)
// transposed==0 -> store out[b][Mout][4096] (natural layout for TDM tiling)
// ---------------------------------------------------------------------------
__global__ __launch_bounds__(256) void proj_gemm_bf16(
    const float* __restrict__ W,    // [Mout, 576] f32
    const float* __restrict__ X,    // [4, 576, 4096] f32
    bf16_t* __restrict__ out,       // see above
    int Mout, int transposed)
{
  const int lane = threadIdx.x & 31;
  const int wave = threadIdx.x >> 5;
  const int b    = blockIdx.z;
  const int mt   = blockIdx.y;
  const int nt   = blockIdx.x * 8 + wave;
  const int l15  = lane & 15;
  const int kb   = (lane >> 4) * 8;
  const int kB   = (lane >> 4) * 16;

  const float* wrow = W + (size_t)(mt * 16 + l15) * 576;
  const float* xcol = X + ((size_t)b * 576) * 4096 + (nt * 16 + l15);

  v8f acc = {};
  for (int kk = 0; kk < 576; kk += 32) {
    v16bf a, bb;
#pragma unroll
    for (int i = 0; i < 8; ++i) {
      a[i]     = (bf16_t)wrow[kk + kb + i];
      a[i + 8] = (bf16_t)wrow[kk + kb + 16 + i];
    }
#pragma unroll
    for (int i = 0; i < 16; ++i)
      bb[i] = (bf16_t)xcol[(size_t)(kk + kb + i + (i >= 8 ? 8 : 0)) * 4096];
    acc = WMMA_BF16(a, bb, acc);
  }
  // D layout: lane&15 = n column; M row = (lane>>4)*8 + v
  if (transposed) {
    v8bf o;
#pragma unroll
    for (int v = 0; v < 8; ++v) o[v] = (bf16_t)acc[v];
    bf16_t* orow = out + ((size_t)b * 4096 + nt * 16 + l15) * Mout + mt * 16 + kb;
    *(v8bf*)orow = o;
  } else {
    bf16_t* obase = out + ((size_t)b * Mout + mt * 16 + kb) * 4096 + nt * 16 + l15;
#pragma unroll
    for (int v = 0; v < 8; ++v) obase[(size_t)v * 4096] = (bf16_t)acc[v];
  }
}

// ---------------------------------------------------------------------------
// Kernel 2: per-query softmax stats (max, sumexp) over all 4096 keys.
// ---------------------------------------------------------------------------
__global__ __launch_bounds__(256) void attn_stats(
    const bf16_t* __restrict__ fT,  // keys^T    [4][4096][64]
    const bf16_t* __restrict__ gT,  // queries^T [4][4096][64]
    float* __restrict__ Mst, float* __restrict__ Lst)
{
  const int lane = threadIdx.x & 31;
  const int wave = threadIdx.x >> 5;
  const int task = blockIdx.x * 8 + wave;
  const int b    = task >> 8;
  const int mt   = task & 255;
  const int l15  = lane & 15;
  const int kb   = (lane >> 4) * 8;
  const int kB   = (lane >> 4) * 16;

  const bf16_t* qrow = gT + ((size_t)b * 4096 + mt * 16 + l15) * 64;
  v16bf q0 = *(const v16bf*)(qrow + kB);
  v16bf q1 = *(const v16bf*)(qrow + 32 + kB);

  float rmax = -INFINITY, rsum = 0.0f;
  const bf16_t* kmat = fT + ((size_t)b * 4096) * 64;
  for (int nt = 0; nt < 256; ++nt) {
    const bf16_t* krow = kmat + (size_t)(nt * 16 + l15) * 64;
    __builtin_prefetch(krow + 16 * 64, 0, 1);
    v16bf a0 = load_afrag(krow, kb);
    v16bf a1 = load_afrag(krow + 32, kb);
    v8f s = {};
    s = WMMA_BF16(a0, q0, s);
    s = WMMA_BF16(a1, q1, s);
    float tmax = s[0];
#pragma unroll
    for (int v = 1; v < 8; ++v) tmax = fmaxf(tmax, s[v]);
    float mnew = fmaxf(rmax, tmax);
    float add = 0.0f;
#pragma unroll
    for (int v = 0; v < 8; ++v) add += __expf(s[v] - mnew);
    rsum = rsum * __expf(rmax - mnew) + add;
    rmax = mnew;
  }
  float om = __shfl_xor(rmax, 16, 32);
  float os = __shfl_xor(rsum, 16, 32);
  float mm = fmaxf(rmax, om);
  float ss = rsum * __expf(rmax - mm) + os * __expf(om - mm);
  if (lane < 16) {
    Mst[(size_t)b * 4096 + mt * 16 + lane] = mm;
    Lst[(size_t)b * 4096 + mt * 16 + lane] = ss;
  }
}

// ---------------------------------------------------------------------------
// Kernel 3: O^T[m, c] = sum_n P[m,n] * h[c, n]; out = gamma*O + x.
// 9 waves/block. Per 32-key iteration:
//   wave 0 issues a TDM load of the [576 c x 32 n] value tile into LDS
//   (TENSORcnt), waves 0..1 compute S + P=exp(s-M)/L into LDS overlapping the
//   DMA, then all 9 waves run 4 PV WMMAs each; B-fragments are contiguous
//   32-byte LDS reads thanks to the [c][32n] tile layout.
// ---------------------------------------------------------------------------
__global__ __launch_bounds__(288) void attn_out(
    const bf16_t* __restrict__ fT, const bf16_t* __restrict__ gT,
    const bf16_t* __restrict__ hN,   // values [4][576][4096]
    const float* __restrict__ Mst, const float* __restrict__ Lst,
    const float* __restrict__ X,     // residual [4][576][4096]
    const float* __restrict__ gammap,
    float* __restrict__ Out)
{
  __shared__ bf16_t Hbuf[576 * 32];  // value tile staged by TDM: [c][32 keys]
  __shared__ bf16_t P[16 * 32];      // probability tile [query m][32 keys]
  const int lane = threadIdx.x & 31;
  const int wave = threadIdx.x >> 5; // 0..8
  const int b    = blockIdx.y;
  const int mt   = blockIdx.x;
  const int l15  = lane & 15;
  const int kb   = (lane >> 4) * 8;
  const int kB   = (lane >> 4) * 16;
  const unsigned hbuf_lds = (unsigned)(uintptr_t)&Hbuf[0];

  v16bf q0 = {}, q1 = {};
  float sM = 0.0f, sLinv = 0.0f;
  if (wave < 2) {
    const bf16_t* qrow = gT + ((size_t)b * 4096 + mt * 16 + l15) * 64;
    q0 = *(const v16bf*)(qrow + kB);
    q1 = *(const v16bf*)(qrow + 32 + kB);
    sM    = Mst[(size_t)b * 4096 + mt * 16 + l15];
    sLinv = 1.0f / Lst[(size_t)b * 4096 + mt * 16 + l15];
  }
  const int c0 = wave * 64;
  v8f acc0 = {}, acc1 = {}, acc2 = {}, acc3 = {};

  for (int it = 0; it < 128; ++it) {
    // Kick the value-tile DMA first so it overlaps the S/P computation.
    if (wave == 0)
      tdm_load_2d_bf16(hbuf_lds, hN + ((size_t)b * 576) * 4096 + it * 32,
                       /*tile_d0=*/32, /*tile_d1=*/576,
                       /*tensor_d0=*/4096, /*tensor_d1=*/576,
                       /*stride0=*/4096ull);
    if (wave < 2) {
      const bf16_t* krow = fT + ((size_t)b * 4096 + it * 32 + wave * 16 + l15) * 64;
      __builtin_prefetch(krow + 32 * 64, 0, 1);
      v16bf a0 = load_afrag(krow, kb);
      v16bf a1 = load_afrag(krow + 32, kb);
      v8f s = {};
      s = WMMA_BF16(a0, q0, s);
      s = WMMA_BF16(a1, q1, s);
      v8bf p;
#pragma unroll
      for (int v = 0; v < 8; ++v) p[v] = (bf16_t)(__expf(s[v] - sM) * sLinv);
      *(v8bf*)&P[l15 * 32 + wave * 16 + kb] = p;
    }
    if (wave == 0)
      __builtin_amdgcn_s_wait_tensorcnt(0);
    __syncthreads();

    // A fragment for PV: row = query m, K = 32 keys from LDS
    v16bf pa;
    {
      v8bf lo = *(const v8bf*)&P[l15 * 32 + kB];
      v8bf hi = *(const v8bf*)&P[l15 * 32 + kB + 8];
#pragma unroll
      for (int i = 0; i < 8; ++i) { pa[i] = lo[i]; pa[i + 8] = hi[i]; }
    }
#pragma unroll
    for (int s4 = 0; s4 < 4; ++s4) {
      const int c = c0 + s4 * 16 + l15;
      v16bf bb = *(const v16bf*)&Hbuf[c * 32 + kB];  // 16 consecutive keys
      if      (s4 == 0) acc0 = WMMA_BF16(pa, bb, acc0);
      else if (s4 == 1) acc1 = WMMA_BF16(pa, bb, acc1);
      else if (s4 == 2) acc2 = WMMA_BF16(pa, bb, acc2);
      else              acc3 = WMMA_BF16(pa, bb, acc3);
    }
    __syncthreads();
  }

  const float gamma = gammap[0];
#pragma unroll
  for (int s4 = 0; s4 < 4; ++s4) {
    v8f a = (s4 == 0) ? acc0 : (s4 == 1) ? acc1 : (s4 == 2) ? acc2 : acc3;
    const int c = c0 + s4 * 16 + l15;
    const size_t base = ((size_t)b * 576 + c) * 4096 + mt * 16 + kb;
#pragma unroll
    for (int v = 0; v < 8; ++v)
      Out[base + v] = gamma * a[v] + X[base + v];
  }
}

// ---------------------------------------------------------------------------
extern "C" void kernel_launch(void* const* d_in, const int* in_sizes, int n_in,
                              void* d_out, int out_size, void* d_ws, size_t ws_size,
                              hipStream_t stream) {
  const float* x     = (const float*)d_in[0];  // [4,576,64,64]
  const float* Wq    = (const float*)d_in[1];  // [64,576]   -> keys f
  const float* Wk    = (const float*)d_in[2];  // [64,576]   -> queries g
  const float* Wv    = (const float*)d_in[3];  // [576,576]  -> values h
  const float* gamma = (const float*)d_in[4];  // [1]
  float* out = (float*)d_out;

  char* ws = (char*)d_ws;
  bf16_t* fT = (bf16_t*)(ws);
  bf16_t* gT = (bf16_t*)(ws + (size_t)(2u << 20));
  bf16_t* hN = (bf16_t*)(ws + (size_t)(4u << 20));
  float*  Ms = (float*)(ws + (size_t)(4u << 20) + 18874368u);
  float*  Ls = Ms + 4 * 4096;

  dim3 blk(256);
  proj_gemm_bf16<<<dim3(32,  4, 4), blk, 0, stream>>>(Wq, x, fT, 64, 1);
  proj_gemm_bf16<<<dim3(32,  4, 4), blk, 0, stream>>>(Wk, x, gT, 64, 1);
  proj_gemm_bf16<<<dim3(32, 36, 4), blk, 0, stream>>>(Wv, x, hN, 576, 0);
  attn_stats<<<dim3(128), blk, 0, stream>>>(fT, gT, Ms, Ls);
  attn_out<<<dim3(256, 4), dim3(288), 0, stream>>>(fT, gT, hN, Ms, Ls, x, gamma, out);
}
